// HR2O_NL_84301618086557
// MI455X (gfx1250) — matrix operand order repaired and checked by reading, not compiled, not run on hardware
//
#include <hip/hip_runtime.h>

typedef __attribute__((ext_vector_type(16))) __bf16 v16bf;
typedef __attribute__((ext_vector_type(8)))  __bf16 v8bf;
typedef __attribute__((ext_vector_type(8)))  float  v8f;

#define Cc   512
#define Nn   512
#define HW   49
#define PADP 81     // 9x9 halo-padded spatial

__device__ __forceinline__ v8f wmma_bf16(v16bf a, v16bf b, v8f c) {
  return __builtin_amdgcn_wmma_f32_16x16x32_bf16(false, a, false, b, (short)0, c, false, false);
}

// A-fragment (16x32 bf16, MxK): lane L holds row m=L%16, K = {g*16..g*16+15}
__device__ __forceinline__ v16bf load_a16(const __bf16* __restrict__ rowp, int c0) {
  v8bf lo = *(const v8bf*)(rowp + c0);
  v8bf hi = *(const v8bf*)(rowp + c0 + 8);
  return __builtin_shufflevector(lo, hi, 0,1,2,3,4,5,6,7,8,9,10,11,12,13,14,15);
}

// CDNA5 async global->LDS copy (16 bytes per lane), tracked by ASYNCcnt.
__device__ __forceinline__ void async_copy16(const void* gp, void* lp) {
  unsigned lds32 = (unsigned)(unsigned long long)lp;  // LDS aperture: low 32 bits
  unsigned long long g64 = (unsigned long long)gp;
  asm volatile("global_load_async_to_lds_b128 %0, %1, off"
               :: "v"(lds32), "v"(g64) : "memory");
}
__device__ __forceinline__ void wait_async0() {
  asm volatile("s_wait_asynccnt 0x0" ::: "memory");
}

// ---------------- prep kernels ----------------

// x (N,C,1,7,7) f32 -> xbp[n][9*9][c] bf16 with zero halo
__global__ __launch_bounds__(256) void xpad_kernel(const float* __restrict__ x,
                                                   __bf16* __restrict__ xbp) {
  int n = blockIdx.x;
  __bf16* dst = xbp + (size_t)n * PADP * Cc;
  const float* xs = x + (size_t)n * Cc * HW;
  for (int t = threadIdx.x; t < PADP * Cc; t += 256) {
    int pix = t / Cc, c = t % Cc;
    int ph = pix / 9, pw = pix % 9;
    float v = 0.f;
    if (ph >= 1 && ph <= 7 && pw >= 1 && pw <= 7)
      v = xs[(size_t)c * HW + (ph - 1) * 7 + (pw - 1)];
    dst[t] = (__bf16)v;
  }
}

// w (C,C,1,3,3) f32 -> wb[tap][co][ci] bf16
__global__ __launch_bounds__(256) void wxform_kernel(const float* __restrict__ w,
                                                     __bf16* __restrict__ wb) {
  size_t total = (size_t)Cc * Cc * 9;
  for (size_t t = (size_t)blockIdx.x * 256 + threadIdx.x; t < total;
       t += (size_t)gridDim.x * 256) {
    size_t co = t / (Cc * 9);
    size_t rem = t % (Cc * 9);
    size_t ci = rem / 9;
    size_t tap = rem % 9;
    wb[(tap * Cc + co) * Cc + ci] = (__bf16)w[t];
  }
}

// ---------------- fused q/k/v conv: implicit GEMM, async-LDS staged B ----------------
// grid: (25088/64, 512/64), block: 128 threads (4 waves).
// Block computes 64M x 64N for q,k,v; B tiles (3 x 64col x 32K bf16 = 12KB)
// double-buffered in LDS via global_load_async_to_lds_b128.
__global__ __launch_bounds__(128) void conv_qkv_kernel(
    const __bf16* __restrict__ xbp,
    const __bf16* __restrict__ wq, const __bf16* __restrict__ wk,
    const __bf16* __restrict__ wv,
    __bf16* __restrict__ qo, __bf16* __restrict__ ko, __bf16* __restrict__ vTo) {
  __shared__ __bf16 bsm[2][3][64 * 32];  // 24 KB
  int tid = threadIdx.x;
  int wave = tid >> 5, lane = tid & 31, g = lane >> 4;
  int mt = blockIdx.x;
  int n0 = blockIdx.y * 64;
  int mrow = mt * 64 + wave * 16 + (lane & 15);
  int ns = mrow / 49, p = mrow % 49;
  int h = p / 7, w = p % 7;
  const __bf16* xrow = xbp + (size_t)ns * PADP * Cc;
  const __bf16* wmat[3] = {wq, wk, wv};

  // stage s in [0,144): tap = s/16, kb = s%16. 768 x 16B chunks, 6 per thread.
  auto stage = [&](int s, int buf) {
    int tap = s >> 4, kb = s & 15;
    for (int ch = tid; ch < 768; ch += 128) {
      int mat = ch >> 8;
      int rem = ch & 255;
      int col = rem >> 2;
      int sub = rem & 3;
      const __bf16* gp =
          wmat[mat] + ((size_t)tap * Cc + (n0 + col)) * Cc + kb * 32 + sub * 8;
      async_copy16(gp, &bsm[buf][mat][col * 32 + sub * 8]);
    }
  };

  v8f accq[4] = {}, acck[4] = {}, accv[4] = {};
  stage(0, 0);
  for (int s = 0; s < 144; ++s) {
    int buf = s & 1;
    wait_async0();
    __syncthreads();
    if (s + 1 < 144) stage(s + 1, buf ^ 1);
    int tap = s >> 4, kb = s & 15;
    int kh = tap / 3, kw = tap % 3;
    const __bf16* arow = xrow + ((h + kh) * 9 + (w + kw)) * Cc;
    v16bf aF = load_a16(arow, kb * 32 + g * 16);
#pragma unroll
    for (int ng = 0; ng < 4; ++ng) {
      int off = (ng * 16 + (lane & 15)) * 32 + g * 16;
      v16bf bq = *(const v16bf*)&bsm[buf][0][off];
      accq[ng] = wmma_bf16(aF, bq, accq[ng]);
      v16bf bk = *(const v16bf*)&bsm[buf][1][off];
      acck[ng] = wmma_bf16(aF, bk, acck[ng]);
      v16bf bv = *(const v16bf*)&bsm[buf][2][off];
      accv[ng] = wmma_bf16(aF, bv, accv[ng]);
    }
  }
  // epilogue: q,k -> [p][i][c]; v -> [p][c][i]
  int mi[8], mp[8];
#pragma unroll
  for (int r = 0; r < 8; ++r) {
    int m = mt * 64 + wave * 16 + r + (g << 3);
    mi[r] = m / 49;
    mp[r] = m % 49;
  }
#pragma unroll
  for (int ng = 0; ng < 4; ++ng) {
    int co = n0 + ng * 16 + (lane & 15);
#pragma unroll
    for (int r = 0; r < 8; ++r) {
      size_t qi = ((size_t)mp[r] * Nn + mi[r]) * Cc + co;
      qo[qi] = (__bf16)accq[ng][r];
      ko[qi] = (__bf16)acck[ng][r];
      size_t vi = ((size_t)mp[r] * Cc + co) * Nn + mi[r];
      vTo[vi] = (__bf16)accv[ng][r];
    }
  }
}

// ---------------- attention per spatial position ----------------
// grid: (49, 512/64), block: 128 threads (4 waves), each wave owns 16 rows.
__global__ __launch_bounds__(128) void attn_kernel(
    const __bf16* __restrict__ q, const __bf16* __restrict__ k,
    const __bf16* __restrict__ vT, const int* __restrict__ rois,
    float* __restrict__ virt) {
  __shared__ float att[64][512];
  __shared__ int grp[512];
  int p = blockIdx.x;
  int ib = blockIdx.y * 64;
  int tid = threadIdx.x;
  int wave = tid >> 5, lane = tid & 31;
  int g = lane >> 4;
  for (int t = tid; t < 512; t += 128) grp[t] = rois[t * 5];
  __syncthreads();

  const __bf16* qp = q + (size_t)p * Nn * Cc;
  const __bf16* kp = k + (size_t)p * Nn * Cc;
  const __bf16* vp = vT + (size_t)p * Cc * Nn;
  const float rscale = 0.04419417382415922f;  // 1/sqrt(512)
  const float NEGINF = -__builtin_inff();

  // phase 1: logits = Q K^T / sqrt(C), masked
  const __bf16* qrow = qp + (size_t)(ib + wave * 16 + (lane & 15)) * Cc;
  for (int jt = 0; jt < 32; ++jt) {
    v8f acc = {};
    int j = jt * 16 + (lane & 15);
    const __bf16* krow = kp + (size_t)j * Cc;
    for (int kb = 0; kb < 16; ++kb) {
      v16bf aF = load_a16(qrow, kb * 32 + g * 16);
      v16bf bF = *(const v16bf*)(krow + kb * 32 + g * 16);
      acc = wmma_bf16(aF, bF, acc);
    }
    int jg = grp[j];
#pragma unroll
    for (int r = 0; r < 8; ++r) {
      int il = wave * 16 + r + (g << 3);
      float v = acc[r] * rscale;
      if (grp[ib + il] != jg) v = NEGINF;
      att[il][j] = v;
    }
  }
  __syncthreads();

  // phase 2: softmax over j (wave32 cooperative, 16 rows per wave)
  for (int r = 0; r < 16; ++r) {
    int il = wave * 16 + r;
    float m = NEGINF;
    for (int t = lane; t < 512; t += 32) m = fmaxf(m, att[il][t]);
    for (int s = 16; s > 0; s >>= 1) m = fmaxf(m, __shfl_xor(m, s, 32));
    float sum = 0.f;
    for (int t = lane; t < 512; t += 32) {
      float e = __expf(att[il][t] - m);
      att[il][t] = e;
      sum += e;
    }
    for (int s = 16; s > 0; s >>= 1) sum += __shfl_xor(sum, s, 32);
    float inv = 1.f / sum;
    for (int t = lane; t < 512; t += 32) att[il][t] *= inv;
  }
  __syncthreads();

  // phase 3: virt = att @ V  (A from LDS, B = vT rows contiguous in j)
  for (int ct = 0; ct < 32; ++ct) {
    v8f acc = {};
    int c = ct * 16 + (lane & 15);
    const __bf16* vrow = vp + (size_t)c * Nn;
    int arow = wave * 16 + (lane & 15);
    for (int kb = 0; kb < 16; ++kb) {
      int j0 = kb * 32 + g * 16;
      v16bf aF;
#pragma unroll
      for (int t = 0; t < 8; ++t) {
        aF[t] = (__bf16)att[arow][j0 + t];
        aF[8 + t] = (__bf16)att[arow][j0 + 8 + t];
      }
      v16bf bF = *(const v16bf*)(vrow + kb * 32 + g * 16);
      acc = wmma_bf16(aF, bF, acc);
    }
#pragma unroll
    for (int r = 0; r < 8; ++r) {
      int i = ib + wave * 16 + r + (g << 3);
      virt[((size_t)i * HW + p) * Cc + c] = acc[r];
    }
  }
}

// ---------------- per-sample mean / rstd ----------------
__global__ __launch_bounds__(256) void stats_kernel(const float* __restrict__ virt,
                                                    float* __restrict__ stats) {
  int i = blockIdx.x;
  const float* v = virt + (size_t)i * HW * Cc;
  float s = 0.f, ss = 0.f;
  for (int t = threadIdx.x; t < HW * Cc; t += 256) {
    float xv = v[t];
    s += xv;
    ss += xv * xv;
  }
  __shared__ float rs[256], rss[256];
  rs[threadIdx.x] = s;
  rss[threadIdx.x] = ss;
  __syncthreads();
  for (int st = 128; st > 0; st >>= 1) {
    if (threadIdx.x < st) {
      rs[threadIdx.x] += rs[threadIdx.x + st];
      rss[threadIdx.x] += rss[threadIdx.x + st];
    }
    __syncthreads();
  }
  if (threadIdx.x == 0) {
    float mean = rs[0] / 25088.f;
    float var = rss[0] / 25088.f - mean * mean;
    stats[i * 2] = mean;
    stats[i * 2 + 1] = rsqrtf(var + 1e-5f);
  }
}

// ---------------- normalize + relu + bf16 + halo pad ----------------
__global__ __launch_bounds__(256) void norm_pad_kernel(
    const float* __restrict__ virt, const float* __restrict__ stats,
    const float* __restrict__ gamma, const float* __restrict__ beta,
    __bf16* __restrict__ vnb) {
  int i = blockIdx.x;
  float mean = stats[i * 2], rstd = stats[i * 2 + 1];
  __bf16* dst = vnb + (size_t)i * PADP * Cc;
  for (int t = threadIdx.x; t < PADP * Cc; t += 256) {
    int pix = t / Cc, c = t % Cc;
    int ph = pix / 9, pw = pix % 9;
    float val = 0.f;
    if (ph >= 1 && ph <= 7 && pw >= 1 && pw <= 7) {
      int p = (ph - 1) * 7 + (pw - 1);
      float xv = virt[((size_t)i * HW + p) * Cc + c];
      xv = (xv - mean) * rstd * gamma[c] + beta[c];
      val = fmaxf(xv, 0.f);
    }
    dst[t] = (__bf16)val;
  }
}

// ---------------- out conv + residual: async-LDS staged B ----------------
// grid: (25088/64, 512/64), block: 128 threads (4 waves).
__global__ __launch_bounds__(128) void conv_out_kernel(
    const __bf16* __restrict__ vnb, const __bf16* __restrict__ wo,
    const float* __restrict__ x, float* __restrict__ out) {
  __shared__ __bf16 bsm[2][64 * 32];  // 8 KB
  int tid = threadIdx.x;
  int wave = tid >> 5, lane = tid & 31, g = lane >> 4;
  int mt = blockIdx.x;
  int n0 = blockIdx.y * 64;
  int mrow = mt * 64 + wave * 16 + (lane & 15);
  int ns = mrow / 49, p = mrow % 49;
  int h = p / 7, w = p % 7;
  const __bf16* base = vnb + (size_t)ns * PADP * Cc;

  auto stage = [&](int s, int buf) {
    int tap = s >> 4, kb = s & 15;
    for (int ch = tid; ch < 256; ch += 128) {
      int col = ch >> 2;
      int sub = ch & 3;
      const __bf16* gp =
          wo + ((size_t)tap * Cc + (n0 + col)) * Cc + kb * 32 + sub * 8;
      async_copy16(gp, &bsm[buf][col * 32 + sub * 8]);
    }
  };

  v8f acc[4] = {};
  stage(0, 0);
  for (int s = 0; s < 144; ++s) {
    int buf = s & 1;
    wait_async0();
    __syncthreads();
    if (s + 1 < 144) stage(s + 1, buf ^ 1);
    int tap = s >> 4, kb = s & 15;
    int kh = tap / 3, kw = tap % 3;
    const __bf16* arow = base + ((h + kh) * 9 + (w + kw)) * Cc;
    v16bf aF = load_a16(arow, kb * 32 + g * 16);
#pragma unroll
    for (int ng = 0; ng < 4; ++ng) {
      int off = (ng * 16 + (lane & 15)) * 32 + g * 16;
      v16bf bF = *(const v16bf*)&bsm[buf][off];
      acc[ng] = wmma_bf16(aF, bF, acc[ng]);
    }
  }
  int mi[8], mp[8];
#pragma unroll
  for (int r = 0; r < 8; ++r) {
    int m = mt * 64 + wave * 16 + r + (g << 3);
    mi[r] = m / 49;
    mp[r] = m % 49;
  }
#pragma unroll
  for (int ng = 0; ng < 4; ++ng) {
    int co = n0 + ng * 16 + (lane & 15);
#pragma unroll
    for (int r = 0; r < 8; ++r) {
      size_t oi = ((size_t)mi[r] * Cc + co) * HW + mp[r];
      out[oi] = x[oi] + acc[ng][r];
    }
  }
}

extern "C" void kernel_launch(void* const* d_in, const int* in_sizes, int n_in,
                              void* d_out, int out_size, void* d_ws, size_t ws_size,
                              hipStream_t stream) {
  const float* x    = (const float*)d_in[0];
  const int*   rois = (const int*)d_in[1];
  const float* w_q  = (const float*)d_in[2];
  const float* w_k  = (const float*)d_in[3];
  const float* w_v  = (const float*)d_in[4];
  const float* w_o  = (const float*)d_in[5];
  const float* gamma = (const float*)d_in[6];
  const float* beta  = (const float*)d_in[7];
  float* out = (float*)d_out;

  char* ws = (char*)d_ws;
  size_t o = 0;
  const size_t szPad  = (size_t)Nn * PADP * Cc * 2;   // 42.5 MB (bf16)
  const size_t szW    = (size_t)9 * Cc * Cc * 2;      // 4.7 MB  (bf16)
  const size_t szQ    = (size_t)HW * Nn * Cc * 2;     // 25.7 MB (bf16)
  const size_t szVirt = (size_t)Nn * HW * Cc * 4;     // 51.4 MB (f32)

  __bf16* xbp = (__bf16*)(ws + o); o += szPad;        // aliased later by vnb
  __bf16* wqb = (__bf16*)(ws + o); o += szW;
  __bf16* wkb = (__bf16*)(ws + o); o += szW;
  __bf16* wvb = (__bf16*)(ws + o); o += szW;
  __bf16* wob = (__bf16*)(ws + o); o += szW;
  __bf16* qb  = (__bf16*)(ws + o); o += szQ;
  __bf16* kb  = (__bf16*)(ws + o); o += szQ;
  __bf16* vTb = (__bf16*)(ws + o); o += szQ;
  float*  virt = (float*)(ws + o); o += szVirt;
  float*  stats = (float*)(ws + o); o += 512 * 2 * sizeof(float);
  __bf16* vnb = xbp;  // reuse xbp region (dead after conv_qkv)

  xpad_kernel<<<dim3(Nn), dim3(256), 0, stream>>>(x, xbp);
  wxform_kernel<<<dim3(512), dim3(256), 0, stream>>>(w_q, wqb);
  wxform_kernel<<<dim3(512), dim3(256), 0, stream>>>(w_k, wkb);
  wxform_kernel<<<dim3(512), dim3(256), 0, stream>>>(w_v, wvb);
  wxform_kernel<<<dim3(512), dim3(256), 0, stream>>>(w_o, wob);

  conv_qkv_kernel<<<dim3((Nn * HW) / 64, Cc / 64), dim3(128), 0, stream>>>(
      xbp, wqb, wkb, wvb, qb, kb, vTb);

  attn_kernel<<<dim3(HW, Nn / 64), dim3(128), 0, stream>>>(qb, kb, vTb, rois, virt);

  stats_kernel<<<dim3(Nn), dim3(256), 0, stream>>>(virt, stats);
  norm_pad_kernel<<<dim3(Nn), dim3(256), 0, stream>>>(virt, stats, gamma, beta, vnb);

  conv_out_kernel<<<dim3((Nn * HW) / 64, Cc / 64), dim3(128), 0, stream>>>(
      vnb, wob, x, out);
}
